// CDIM_19567871000743
// MI455X (gfx1250) — matrix operand order repaired and checked by reading, not compiled, not run on hardware
//
#include <hip/hip_runtime.h>
#include <hip/hip_bf16.h>

// ---------------------------------------------------------------------------
// Types for WMMA fragments (wave32, V_WMMA_F32_16X16X32_F16)
// ---------------------------------------------------------------------------
typedef _Float16 h8_t  __attribute__((ext_vector_type(8)));
typedef _Float16 h16_t __attribute__((ext_vector_type(16)));
typedef float    f8_t  __attribute__((ext_vector_type(8)));
typedef unsigned int u32x4 __attribute__((ext_vector_type(4)));
typedef int          i32x4 __attribute__((ext_vector_type(4)));
typedef int          i32x8 __attribute__((ext_vector_type(8)));

static __device__ inline h16_t cat8(h8_t lo, h8_t hi) {
    h16_t r;
#pragma unroll
    for (int i = 0; i < 8; ++i) { r[i] = lo[i]; r[8 + i] = hi[i]; }
    return r;
}
static __device__ inline f8_t zero8f() {
    f8_t r;
#pragma unroll
    for (int i = 0; i < 8; ++i) r[i] = 0.0f;
    return r;
}
static __device__ inline f8_t wmma_f32_16x16x32(h16_t a, h16_t b, f8_t c) {
    return __builtin_amdgcn_wmma_f32_16x16x32_f16(false, a, false, b, (short)0, c,
                                                  false, false);
}

// LDS row pitch (halfs) for staged B panels: 64 data + 8 pad to spread banks.
#define BPITCH 72

// ---------------------------------------------------------------------------
// Tensor Data Mover: stage a [rows x 64 halfs] panel (row stride = strideHalfs)
// from global into LDS, inserting a 16B pad after each 128B row so the LDS
// pitch is BPITCH halfs. D# layout per CDNA5 ISA 8.3/8.4 (2D, groups 2/3 = 0).
// ---------------------------------------------------------------------------
#if defined(__has_builtin)
#  if __has_builtin(__builtin_amdgcn_tensor_load_to_lds) && \
      __has_builtin(__builtin_amdgcn_s_wait_tensorcnt)
#    define USE_TDM 1
#  endif
#endif
#ifndef USE_TDM
#  define USE_TDM 0
#endif

#if USE_TDM
static __device__ inline void tdm_load_panel(const _Float16* gsrc, unsigned lds_off,
                                             int rows, long long strideHalfs) {
    unsigned long long ga = (unsigned long long)gsrc;
    u32x4 g0;
    g0[0] = 1u;                                       // count=1, user mode
    g0[1] = lds_off;                                  // lds_addr (bytes)
    g0[2] = (unsigned)(ga & 0xFFFFFFFFu);             // global_addr[31:0]
    g0[3] = (unsigned)((ga >> 32) & 0x01FFFFFFu)      // global_addr[56:32]
          | (2u << 30);                               // type = 2 (image)
    i32x8 g1;
    g1[0] = (1 << 16)        // data_size = 1 -> 2 bytes
          | (1 << 20)        // pad_enable
          | (4 << 22)        // pad_interval: 32 DWORDs (128B) between pads
          | (3 << 25);       // pad_amount: 4 DWORDs (16B) of padding
    g1[1] = 64 << 16;                                 // tensor_dim0 = 64 (lo16)
    g1[2] = (rows & 0xFFFF) << 16;                    // dim0 hi=0 | tensor_dim1 lo16
    g1[3] = 64 << 16;                                 // dim1 hi=0 | tile_dim0 = 64
    g1[4] = rows & 0xFFFF;                            // tile_dim1 = rows, tile_dim2=0
    g1[5] = (int)(strideHalfs & 0xFFFFFFFFll);        // tensor_dim0_stride lo32
    g1[6] = (int)((strideHalfs >> 32) & 0xFFFF);      // stride hi16 | dim1_stride lo16
    g1[7] = 0;                                        // dim1_stride hi32 (2D: unused)
    i32x4 z4;
    z4[0] = 0; z4[1] = 0; z4[2] = 0; z4[3] = 0;
#if __clang_major__ >= 23
    i32x8 z8;
#pragma unroll
    for (int i = 0; i < 8; ++i) z8[i] = 0;
    __builtin_amdgcn_tensor_load_to_lds(g0, g1, z4, z4, z8, 0);
#else
    __builtin_amdgcn_tensor_load_to_lds(g0, g1, z4, z4, 0);
#endif
}
#endif

// ---------------------------------------------------------------------------
// Bicubic 8x downsample: x [B,64,256,256] f32 NCHW -> [B,32,32,64] f16 NHWC
// Exact scale 8, align_corners=False: frac == 0.5 always ->
// constant Keys(a=-0.5) taps [-1/16, 9/16, 9/16, -1/16] at rows 8o+2..8o+5.
// ---------------------------------------------------------------------------
__global__ void k_down8(const float* __restrict__ x, _Float16* __restrict__ o,
                        int total) {
    int idx = blockIdx.x * 256 + threadIdx.x;
    if (idx >= total) return;
    int c = idx & 63;
    int t = idx >> 6;
    int ow = t & 31;
    int oh = (t >> 5) & 31;
    int b  = t >> 10;
    const float wgt[4] = {-0.0625f, 0.5625f, 0.5625f, -0.0625f};
    const float* base = x + (((size_t)b * 64 + c) * 256 + (oh * 8 + 2)) * 256 + (ow * 8 + 2);
    float acc = 0.0f;
#pragma unroll
    for (int i = 0; i < 4; ++i) {
        float rs = 0.0f;
#pragma unroll
        for (int j = 0; j < 4; ++j) rs += wgt[j] * base[i * 256 + j];
        acc += wgt[i] * rs;
    }
    o[idx] = (_Float16)acc;
}

// ---------------------------------------------------------------------------
// Weight prep: [O,I,3,3] f32 -> [9, I, O] f16  (B-matrix friendly)
// ---------------------------------------------------------------------------
__global__ void k_prep_w(const float* __restrict__ w, _Float16* __restrict__ o,
                         int O, int I) {
    int idx = blockIdx.x * 256 + threadIdx.x;
    int total = O * I * 9;
    if (idx >= total) return;
    int kk = idx % 9;
    int rem = idx / 9;
    int i  = rem % I;
    int oc = rem / I;
    o[((size_t)kk * I + i) * O + oc] = (_Float16)w[idx];
}

__global__ void k_zeroh(_Float16* __restrict__ p, int n) {
    int i = blockIdx.x * 256 + threadIdx.x;
    if (i < n) p[i] = (_Float16)0.0f;
}

// ---------------------------------------------------------------------------
// Implicit-GEMM 3x3 SAME conv via WMMA.
//   in  : NHWC f16 [B,H,W,Cin]   (W % 32 == 0)
//   w9  : [9, Cin, 64] f16
//   zrow: Cin halfs of zeros (OOB lanes read here -> no exec masking)
//   bias: [64] f32, ReLU applied
//   outH: NHWC f16 with channel stride coutStride, channels [coutOff,+64)
//   outF: optional NCHW f32 [64,H,W] (single batch)
// One wave computes 32 pixels x 64 Cout (8 accumulators); 4 waves/block.
// Per-tap weight panel staged in LDS via TDM (fallback: cooperative loads).
// ---------------------------------------------------------------------------
__global__ void k_conv3x3(const _Float16* __restrict__ in, int H, int W, int Cin,
                          const _Float16* __restrict__ w9,
                          const _Float16* __restrict__ zrow,
                          const float* __restrict__ bias,
                          _Float16* __restrict__ outH, int coutStride, int coutOff,
                          float* __restrict__ outF) {
    extern __shared__ _Float16 smw[];   // [Cin][BPITCH]
    const int tid  = threadIdx.x;
    const int lane = tid & 31;
    const int wave = tid >> 5;
    const int pbase = (blockIdx.x * 4 + wave) * 32;  // 32 consecutive pixels, one row
    const int w0 = pbase % W;
    const int hh = (pbase / W) % H;
    const int bb = pbase / (W * H);
    const int m  = lane & 15;
    const int kb = (lane >> 4) * 8;

    f8_t acc[8];
#pragma unroll
    for (int t = 0; t < 8; ++t) acc[t] = zero8f();

    for (int tap = 0; tap < 9; ++tap) {
        const _Float16* wtap = w9 + (size_t)tap * Cin * 64;
        __syncthreads();                       // previous tap's readers done
#if USE_TDM
        if (wave == 0) {
            tdm_load_panel(wtap, (unsigned)(unsigned long long)smw, Cin, 64);
            __builtin_amdgcn_s_wait_tensorcnt((short)0);
        }
#else
        for (int i = 0; i < (Cin >> 4); ++i) {
            int idx = tid + i * 128;           // v8h chunk id
            int r = idx >> 3, g = idx & 7;
            *(h8_t*)(smw + r * BPITCH + g * 8) = *(const h8_t*)(wtap + r * 64 + g * 8);
        }
#endif
        __syncthreads();
        if (tap < 8) {  // pull next tap toward L2 while we compute
            const _Float16* nxt = w9 + (size_t)(tap + 1) * Cin * 64;
            __builtin_prefetch(nxt + (size_t)tid * (Cin >> 1), 0, 2);
        }
        const int dy = tap / 3 - 1;
        const int dx = tap % 3 - 1;
        const int hy = hh + dy;
        const int wx0 = w0 + m + dx;
        const int wx1 = wx0 + 16;
        const bool oky = (hy >= 0) && (hy < H);
        const bool in0 = oky && (wx0 >= 0) && (wx0 < W);
        const bool in1 = oky && (wx1 < W);
        const _Float16* ar0 =
            in0 ? in + (((size_t)bb * H + hy) * W + wx0) * Cin : zrow;
        const _Float16* ar1 =
            in1 ? in + (((size_t)bb * H + hy) * W + wx1) * Cin : zrow;
        for (int kc = 0; kc < Cin; kc += 32) {
            h16_t af0 = cat8(*(const h8_t*)(ar0 + kc + kb),
                             *(const h8_t*)(ar0 + kc + kb + 16));
            h16_t af1 = cat8(*(const h8_t*)(ar1 + kc + kb),
                             *(const h8_t*)(ar1 + kc + kb + 16));
            const _Float16* srow = smw + (size_t)(kc + kb) * BPITCH + m;
            h16_t bf[4];
#pragma unroll
            for (int nt = 0; nt < 4; ++nt)
#pragma unroll
                for (int s = 0; s < 8; ++s) {
                    bf[nt][s] = srow[s * BPITCH + nt * 16];
                    bf[nt][8 + s] = srow[(16 + s) * BPITCH + nt * 16];
                }
#pragma unroll
            for (int nt = 0; nt < 4; ++nt)
                acc[nt] = wmma_f32_16x16x32(af0, bf[nt], acc[nt]);
#pragma unroll
            for (int nt = 0; nt < 4; ++nt)
                acc[4 + nt] = wmma_f32_16x16x32(af1, bf[nt], acc[4 + nt]);
        }
    }

    const int n_ = lane & 15;
#pragma unroll
    for (int nt = 0; nt < 4; ++nt) {
        int n = nt * 16 + n_;
        float bv = bias[n];
#pragma unroll
        for (int v = 0; v < 8; ++v) {
            int mr = v + 8 * (lane >> 4);
            float v0 = acc[nt][v] + bv;
            float v1 = acc[4 + nt][v] + bv;
            v0 = v0 > 0.0f ? v0 : 0.0f;
            v1 = v1 > 0.0f ? v1 : 0.0f;
            int wc0 = w0 + mr;
            int wc1 = wc0 + 16;
            size_t pix0 = ((size_t)bb * H + hh) * W + wc0;
            if (outH) {
                outH[pix0 * coutStride + coutOff + n] = (_Float16)v0;
                outH[(pix0 + 16) * coutStride + coutOff + n] = (_Float16)v1;
            }
            if (outF) {
                outF[((size_t)n * H + hh) * W + wc0] = v0;
                outF[((size_t)n * H + hh) * W + wc1] = v1;
            }
        }
    }
}

// ---------------------------------------------------------------------------
// Generic f16 GEMM (single batch): C[M,N] f32 = A[M,K] * B[K,N], row-major.
// Wave tile 32(M) x 64(N); 4 waves along M -> block tile 128 x 64.
// grid.x = M/128, grid.y = N/64. K multiple of 64.
// B panel (64 rows x 64 cols) staged in LDS via TDM (fallback: cooperative).
// ---------------------------------------------------------------------------
__global__ void k_gemm(const _Float16* __restrict__ A, int lda,
                       const _Float16* __restrict__ B, int ldb,
                       float* __restrict__ C, int ldc, int K) {
    __shared__ _Float16 smb[64 * BPITCH];
    const int tid  = threadIdx.x;
    const int lane = tid & 31;
    const int wave = tid >> 5;
    const int mbase = (blockIdx.x * 4 + wave) * 32;
    const int nbase = blockIdx.y * 64;
    const int m  = lane & 15;
    const int kb = (lane >> 4) * 8;

    f8_t acc[8];
#pragma unroll
    for (int t = 0; t < 8; ++t) acc[t] = zero8f();

    const _Float16* arow0 = A + (size_t)(mbase + m) * lda;
    const _Float16* arow1 = arow0 + (size_t)16 * lda;

    for (int k0 = 0; k0 < K; k0 += 64) {
        __syncthreads();
#if USE_TDM
        if (wave == 0) {
            tdm_load_panel(B + (size_t)k0 * ldb + nbase,
                           (unsigned)(unsigned long long)smb, 64, ldb);
            __builtin_amdgcn_s_wait_tensorcnt((short)0);
        }
#else
#pragma unroll
        for (int i = 0; i < 4; ++i) {
            int idx = tid + i * 128;
            int r = idx >> 3, g = idx & 7;
            *(h8_t*)(smb + r * BPITCH + g * 8) =
                *(const h8_t*)(B + (size_t)(k0 + r) * ldb + nbase + g * 8);
        }
#endif
        __syncthreads();
        if (k0 + 64 < K) {  // prefetch next B panel
            const _Float16* nb =
                B + (size_t)(k0 + 64 + (tid >> 1)) * ldb + nbase + (tid & 1) * 32;
            __builtin_prefetch(nb, 0, 2);
        }
#pragma unroll
        for (int kk = 0; kk < 64; kk += 32) {
            h16_t af0 = cat8(*(const h8_t*)(arow0 + k0 + kk + kb),
                             *(const h8_t*)(arow0 + k0 + kk + kb + 16));
            h16_t af1 = cat8(*(const h8_t*)(arow1 + k0 + kk + kb),
                             *(const h8_t*)(arow1 + k0 + kk + kb + 16));
            const _Float16* srow = smb + (size_t)(kk + kb) * BPITCH + m;
            h16_t bf[4];
#pragma unroll
            for (int nt = 0; nt < 4; ++nt)
#pragma unroll
                for (int s = 0; s < 8; ++s) {
                    bf[nt][s] = srow[s * BPITCH + nt * 16];
                    bf[nt][8 + s] = srow[(16 + s) * BPITCH + nt * 16];
                }
#pragma unroll
            for (int nt = 0; nt < 4; ++nt)
                acc[nt] = wmma_f32_16x16x32(af0, bf[nt], acc[nt]);
#pragma unroll
            for (int nt = 0; nt < 4; ++nt)
                acc[4 + nt] = wmma_f32_16x16x32(af1, bf[nt], acc[4 + nt]);
        }
    }

    const int n_ = lane & 15;
#pragma unroll
    for (int nt = 0; nt < 4; ++nt) {
#pragma unroll
        for (int v = 0; v < 8; ++v) {
            int mr = mbase + v + 8 * (lane >> 4);
            C[(size_t)mr * ldc + nbase + nt * 16 + n_] = acc[nt][v];
            C[(size_t)(mr + 16) * ldc + nbase + nt * 16 + n_] = acc[4 + nt][v];
        }
    }
}

// ---------------------------------------------------------------------------
// Row softmax: logits f32 [rows, N] -> mask f16 [rows, N]; one block per row.
// ---------------------------------------------------------------------------
__global__ void k_softmax(const float* __restrict__ L, _Float16* __restrict__ M,
                          int N) {
    const int row = blockIdx.x;
    const float* r = L + (size_t)row * N;
    __shared__ float red[256];
    float mx = -1e30f;
    for (int i = threadIdx.x; i < N; i += 256) mx = fmaxf(mx, r[i]);
    red[threadIdx.x] = mx;
    __syncthreads();
    for (int s = 128; s > 0; s >>= 1) {
        if (threadIdx.x < s) red[threadIdx.x] = fmaxf(red[threadIdx.x], red[threadIdx.x + s]);
        __syncthreads();
    }
    mx = red[0];
    __syncthreads();
    float sum = 0.0f;
    for (int i = threadIdx.x; i < N; i += 256) sum += __expf(r[i] - mx);
    red[threadIdx.x] = sum;
    __syncthreads();
    for (int s = 128; s > 0; s >>= 1) {
        if (threadIdx.x < s) red[threadIdx.x] += red[threadIdx.x + s];
        __syncthreads();
    }
    float inv = 1.0f / red[0];
    for (int i = threadIdx.x; i < N; i += 256)
        M[(size_t)row * N + i] = (_Float16)(__expf(r[i] - mx) * inv);
}

// ---------------------------------------------------------------------------
// Transpose [S,C] -> [C,S] f16, grid.y = batch
// ---------------------------------------------------------------------------
__global__ void k_transpose_sc(const _Float16* __restrict__ in,
                               _Float16* __restrict__ out, int S, int C) {
    int idx = blockIdx.x * 256 + threadIdx.x;
    if (idx >= S * C) return;
    const _Float16* ib = in + (size_t)blockIdx.y * S * C;
    _Float16* ob = out + (size_t)blockIdx.y * S * C;
    int s = idx / C, c = idx % C;
    ob[(size_t)c * S + s] = ib[idx];
}

__global__ void k_addh(const _Float16* __restrict__ a, const _Float16* __restrict__ b,
                       _Float16* __restrict__ o, int n) {
    int i = blockIdx.x * 256 + threadIdx.x;
    if (i < n) o[i] = (_Float16)((float)a[i] + (float)b[i]);
}

// ---------------------------------------------------------------------------
// Bicubic 8x upsample of refine [32,32,64] f32 (NHWC) * gamma + orig (NCHW f32)
// -> f16 channel slice of cat1 [256,256,256]
// ---------------------------------------------------------------------------
static __device__ inline float cubw(float t) {  // Keys a = -0.5
    float at = fabsf(t);
    if (at <= 1.0f) return ((1.5f * at - 2.5f) * at) * at + 1.0f;
    if (at < 2.0f)  return ((-0.5f * at + 2.5f) * at - 4.0f) * at + 2.0f;
    return 0.0f;
}

__global__ void k_upsample_add(const float* __restrict__ refine,
                               const float* __restrict__ orig,
                               const float* __restrict__ gamma,
                               _Float16* __restrict__ out, int choff) {
    int idx = blockIdx.x * 256 + threadIdx.x;  // 256*256*64
    if (idx >= 256 * 256 * 64) return;
    int c = idx & 63;
    int t = idx >> 6;
    int ow = t & 255;
    int oh = t >> 8;
    float g = gamma[0];
    float fy = (oh + 0.5f) * 0.125f - 0.5f;
    float fx = (ow + 0.5f) * 0.125f - 0.5f;
    int y0 = (int)floorf(fy);
    int x0 = (int)floorf(fx);
    float ty = fy - (float)y0;
    float tx = fx - (float)x0;
    float wy[4], wx[4];
#pragma unroll
    for (int i = 0; i < 4; ++i) {
        wy[i] = cubw(ty - (float)(i - 1));
        wx[i] = cubw(tx - (float)(i - 1));
    }
    float acc = 0.0f;
#pragma unroll
    for (int i = 0; i < 4; ++i) {
        int yy = y0 - 1 + i;
        yy = yy < 0 ? 0 : (yy > 31 ? 31 : yy);
        float rs = 0.0f;
#pragma unroll
        for (int j = 0; j < 4; ++j) {
            int xx = x0 - 1 + j;
            xx = xx < 0 ? 0 : (xx > 31 ? 31 : xx);
            rs += wx[j] * refine[((yy * 32 + xx) << 6) + c];
        }
        acc += wy[i] * rs;
    }
    float val = g * acc + orig[((size_t)c * 256 + oh) * 256 + ow];
    out[(size_t)t * 256 + choff + c] = (_Float16)val;
}

// ---------------------------------------------------------------------------
// Spatial attention: channel mean/max, 3x3 conv(2->1), sigmoid,
// out = (1 + sigmoid) * x, written into cat2 channel slice (f16 NHWC).
// ---------------------------------------------------------------------------
__global__ void k_avgmax(const float* __restrict__ x, float* __restrict__ am) {
    int p = blockIdx.x * 256 + threadIdx.x;
    if (p >= 65536) return;
    float s = 0.0f, mx = -1e30f;
    for (int c = 0; c < 64; ++c) {
        float v = x[(size_t)c * 65536 + p];
        s += v;
        mx = fmaxf(mx, v);
    }
    am[p] = s * (1.0f / 64.0f);
    am[65536 + p] = mx;
}

__global__ void k_sa_apply(const float* __restrict__ x, const float* __restrict__ am,
                           const float* __restrict__ wsa,
                           _Float16* __restrict__ out, int choff) {
    int p = blockIdx.x * 256 + threadIdx.x;
    if (p >= 65536) return;
    int ow = p & 255, oh = p >> 8;
    float a = 0.0f;
#pragma unroll
    for (int ch = 0; ch < 2; ++ch)
#pragma unroll
        for (int ky = 0; ky < 3; ++ky)
#pragma unroll
            for (int kx = 0; kx < 3; ++kx) {
                int yy = oh + ky - 1, xx = ow + kx - 1;
                if (yy >= 0 && yy < 256 && xx >= 0 && xx < 256)
                    a += wsa[(ch * 3 + ky) * 3 + kx] * am[ch * 65536 + yy * 256 + xx];
            }
    float s = 1.0f / (1.0f + __expf(-a));
    float sc = 1.0f + s;
    for (int c = 0; c < 64; ++c)
        out[(size_t)p * 192 + choff + c] = (_Float16)(sc * x[(size_t)c * 65536 + p]);
}

// ---------------------------------------------------------------------------
// Host launcher
// ---------------------------------------------------------------------------
extern "C" void kernel_launch(void* const* d_in, const int* in_sizes, int n_in,
                              void* d_out, int out_size, void* d_ws, size_t ws_size,
                              hipStream_t stream) {
    (void)in_sizes; (void)n_in; (void)out_size; (void)ws_size;
    const float* x = (const float*)d_in[0];
    const float* y = (const float*)d_in[1];
    const float* w_qkv[6];
    const float* b_qkv[6];
    for (int i = 0; i < 6; ++i) {
        w_qkv[i] = (const float*)d_in[2 + 2 * i];
        b_qkv[i] = (const float*)d_in[3 + 2 * i];
    }
    const float* w_reduce = (const float*)d_in[14];
    const float* b_reduce = (const float*)d_in[15];
    const float* w_sec    = (const float*)d_in[16];
    const float* b_sec    = (const float*)d_in[17];
    const float* w_sa_rgb = (const float*)d_in[18];
    const float* w_sa_inf = (const float*)d_in[19];
    const float* gamma[4] = {(const float*)d_in[20], (const float*)d_in[21],
                             (const float*)d_in[22], (const float*)d_in[23]};
    float* out = (float*)d_out;

    const int Bn = 4, C = 64, Hs = 32, Ws = 32, S = 1024, Hf = 256, Wf = 256;
    const int P = 65536;  // 256*256

    char* wsp = (char*)d_ws;
    auto alloc = [&](size_t bytes) -> void* {
        void* p = (void*)wsp;
        wsp += (bytes + 255) & ~(size_t)255;
        return p;
    };
    _Float16* x32 = (_Float16*)alloc((size_t)Bn * S * C * 2);
    _Float16* y32 = (_Float16*)alloc((size_t)Bn * S * C * 2);
    _Float16* Qr = (_Float16*)alloc((size_t)Bn * S * C * 2);
    _Float16* Kr = (_Float16*)alloc((size_t)Bn * S * C * 2);
    _Float16* Vr = (_Float16*)alloc((size_t)Bn * S * C * 2);
    _Float16* Qi = (_Float16*)alloc((size_t)Bn * S * C * 2);
    _Float16* Ki = (_Float16*)alloc((size_t)Bn * S * C * 2);
    _Float16* Vi = (_Float16*)alloc((size_t)Bn * S * C * 2);
    _Float16* Qrt = (_Float16*)alloc((size_t)Bn * C * S * 2);
    _Float16* Qit = (_Float16*)alloc((size_t)Bn * C * S * 2);
    _Float16* Vd  = (_Float16*)alloc((size_t)Bn * S * C * 2);
    _Float16* w9q[6];
    for (int i = 0; i < 6; ++i) w9q[i] = (_Float16*)alloc((size_t)9 * 64 * 64 * 2);
    _Float16* w9red = (_Float16*)alloc((size_t)9 * 256 * 64 * 2);
    _Float16* w9sec = (_Float16*)alloc((size_t)9 * 192 * 64 * 2);
    _Float16* zrow  = (_Float16*)alloc((size_t)256 * 2);       // zero row (Cin<=256)
    float*    logits = (float*)alloc((size_t)S * S * 4);
    _Float16* maskb  = (_Float16*)alloc((size_t)S * S * 2);
    float*    refine = (float*)alloc((size_t)S * C * 4);
    float*    avgmax = (float*)alloc((size_t)2 * P * 4);
    _Float16* cat1   = (_Float16*)alloc((size_t)P * 256 * 2);  // r1..r4 concat (NHWC)
    _Float16* cat2   = (_Float16*)alloc((size_t)P * 192 * 2);  // glob|sa_inf|sa_rgb

    // ---- weight preprocessing + zero row --------------------------------
    for (int i = 0; i < 6; ++i)
        k_prep_w<<<(64 * 64 * 9 + 255) / 256, 256, 0, stream>>>(w_qkv[i], w9q[i], 64, 64);
    k_prep_w<<<(64 * 256 * 9 + 255) / 256, 256, 0, stream>>>(w_reduce, w9red, 64, 256);
    k_prep_w<<<(64 * 192 * 9 + 255) / 256, 256, 0, stream>>>(w_sec, w9sec, 64, 192);
    k_zeroh<<<1, 256, 0, stream>>>(zrow, 256);

    // ---- 8x bicubic downsample to 32x32 ---------------------------------
    int nds = Bn * S * C;
    k_down8<<<(nds + 255) / 256, 256, 0, stream>>>(x, x32, nds);
    k_down8<<<(nds + 255) / 256, 256, 0, stream>>>(y, y32, nds);

    // ---- QKV convs (B=4, 32x32, Cin=64) via WMMA ------------------------
    _Float16* qkvOut[6] = {Qr, Kr, Vr, Qi, Ki, Vi};
    const _Float16* qkvIn[6] = {x32, x32, x32, y32, y32, y32};
    int pxS = Bn * Hs * Ws;  // 4096
    for (int i = 0; i < 6; ++i)
        k_conv3x3<<<pxS / 128, 128, (size_t)64 * BPITCH * 2, stream>>>(
            qkvIn[i], Hs, Ws, 64, w9q[i], zrow, b_qkv[i], qkvOut[i], 64, 0, nullptr);

    k_addh<<<(nds + 255) / 256, 256, 0, stream>>>(Vr, Vi, Vd, nds);
    k_transpose_sc<<<dim3(S * C / 256, Bn), 256, 0, stream>>>(Qr, Qrt, S, C);
    k_transpose_sc<<<dim3(S * C / 256, Bn), 256, 0, stream>>>(Qi, Qit, S, C);

    // ---- per-batch attention + fusion pipeline --------------------------
    const _Float16* Qt_[4] = {Qrt, Qit, Qrt, Qit};
    const _Float16* Km_[4] = {Kr, Ki, Ki, Kr};
    const _Float16* Vm_[4] = {Vd, Vd, Vr, Vi};
    const float* orig_[4]  = {x, y, y, x};

    for (int b = 0; b < Bn; ++b) {
        for (int a = 0; a < 4; ++a) {
            // logits[s][t] = sum_c K[s][c] * Qt[c][t]   (M=N=S, K=64)
            k_gemm<<<dim3(S / 128, S / 64, 1), 128, 0, stream>>>(
                Km_[a] + (size_t)b * S * C, 64,
                Qt_[a] + (size_t)b * C * S, S,
                logits, S, 64);
            k_softmax<<<S, 256, 0, stream>>>(logits, maskb, S);
            // refine[t][c] = sum_s mask[t][s] * V[s][c]  (M=S, N=64, K=S)
            k_gemm<<<dim3(S / 128, 1, 1), 128, 0, stream>>>(
                maskb, S,
                Vm_[a] + (size_t)b * S * C, 64,
                refine, 64, S);
            k_upsample_add<<<(P * 64) / 256, 256, 0, stream>>>(
                refine, orig_[a] + (size_t)b * C * P, gamma[a], cat1, a * 64);
        }
        // glob = relu(conv3x3(cat1, Cin=256) + b_reduce) -> cat2 channels 0..63
        k_conv3x3<<<P / 128, 128, (size_t)256 * BPITCH * 2, stream>>>(
            cat1, Hf, Wf, 256, w9red, zrow, b_reduce, cat2, 192, 0, nullptr);
        // sa_inf -> channels 64..127 ; sa_rgb -> channels 128..191
        k_avgmax<<<P / 256, 256, 0, stream>>>(y + (size_t)b * C * P, avgmax);
        k_sa_apply<<<P / 256, 256, 0, stream>>>(y + (size_t)b * C * P, avgmax,
                                                w_sa_inf, cat2, 64);
        k_avgmax<<<P / 256, 256, 0, stream>>>(x + (size_t)b * C * P, avgmax);
        k_sa_apply<<<P / 256, 256, 0, stream>>>(x + (size_t)b * C * P, avgmax,
                                                w_sa_rgb, cat2, 128);
        // out = relu(conv3x3(cat2, Cin=192) + b_sec) -> f32 NCHW d_out
        k_conv3x3<<<P / 128, 128, (size_t)192 * BPITCH * 2, stream>>>(
            cat2, Hf, Wf, 192, w9sec, zrow, b_sec, nullptr, 0, 0,
            out + (size_t)b * C * P);
    }
}
